// QuantizedCONV_66975720013864
// MI455X (gfx1250) — 8-bit matrix operand order repaired and checked by reading, compiled, not run on hardware
//
#include <hip/hip_runtime.h>
#include <stdint.h>
#include <stddef.h>

typedef __attribute__((ext_vector_type(8))) int v8i;

#define IDX_ABSMAX_X 0
#define IDX_MAX1     1
#define IDX_MAX2     2
#define IDX_MAX3     3
#define IDX_A0       4
#define IDX_W1S      5
#define IDX_W2S      6
#define IDX_FW1S     7
#define IDX_FW2S     8

#define OFF_SCAL   0
#define OFF_W1Q    2048
#define OFF_W2Q    4096
#define OFF_FW1Q   8192
#define OFF_FW2Q   65536
#define ZERO_BYTES 131072

#define OFF_A      131072
#define OFF_XQ     (OFF_A)
#define OFF_ACT1Q  (OFF_A)
#define OFF_ACT2Q  (OFF_A + 16777216)
#define OFF_FC1Q   (OFF_A + 20971520)
#define OFF_B      (OFF_A + 26214400)
#define OFF_POOL1F (OFF_B)
#define OFF_POOL2F (OFF_B)
#define OFF_FC1F   (OFF_B)

static __device__ __forceinline__ void atomic_max_pos(float* addr, float v) {
  atomicMax(reinterpret_cast<unsigned int*>(addr), __float_as_uint(v));
}
static __device__ __forceinline__ float wave_max(float v) {
  for (int off = 16; off > 0; off >>= 1) v = fmaxf(v, __shfl_down(v, off, 32));
  return v;
}
static __device__ __forceinline__ float bias_q(float b, float s) {
  float r = rintf(b / s);
  return fminf(fmaxf(r, -2147483648.0f), 2147483647.0f);
}

__global__ void k_zero(float* p, int n) {
  int i = blockIdx.x * blockDim.x + threadIdx.x;
  if (i < n) p[i] = 0.0f;
}

__global__ void k_absmax(const float* __restrict__ x, int n, float* __restrict__ out) {
  float v = 0.0f;
  for (int i = blockIdx.x * blockDim.x + threadIdx.x; i < n; i += gridDim.x * blockDim.x)
    v = fmaxf(v, fabsf(x[i]));
  v = wave_max(v);
  if ((threadIdx.x & 31) == 0) atomic_max_pos(out, v);
}

__global__ void k_quant_weights(const float* __restrict__ w1, const float* __restrict__ w2,
                                const float* __restrict__ fw1, const float* __restrict__ fw2,
                                float* __restrict__ scal,
                                int8_t* __restrict__ w1q, int8_t* __restrict__ w2q,
                                int8_t* __restrict__ fw1q, int8_t* __restrict__ fw2q) {
  __shared__ float red[256];
  const int t = threadIdx.x;
  auto block_absmax = [&](const float* p, int n) -> float {
    float v = 0.0f;
    for (int i = t; i < n; i += 256) v = fmaxf(v, fabsf(p[i]));
    red[t] = v; __syncthreads();
    for (int s = 128; s > 0; s >>= 1) { if (t < s) red[t] = fmaxf(red[t], red[t + s]); __syncthreads(); }
    float r = red[0]; __syncthreads(); return r;
  };
  const float w1s  = block_absmax(w1, 6 * 75)     / 127.0f + 1e-12f;
  const float w2s  = block_absmax(w2, 16 * 150)   / 127.0f + 1e-12f;
  const float fw1s = block_absmax(fw1, 120 * 400) / 127.0f + 1e-12f;
  const float fw2s = block_absmax(fw2, 10 * 120)  / 127.0f + 1e-12f;
  const float a0   = scal[IDX_ABSMAX_X] / 127.0f + 1e-12f;
  if (t == 0) {
    scal[IDX_W1S] = w1s; scal[IDX_W2S] = w2s;
    scal[IDX_FW1S] = fw1s; scal[IDX_FW2S] = fw2s;
    scal[IDX_A0] = a0;
  }
  auto q8 = [](float x, float s) -> int8_t {
    float q = fminf(fmaxf(rintf(x / s), -128.0f), 127.0f);
    return (int8_t)(int)q;
  };
  for (int i = t; i < 6 * 75;   i += 256) w1q[i] = q8(w1[i], w1s);
  for (int i = t; i < 16 * 150; i += 256) w2q[i] = q8(w2[i], w2s);
  for (int i = t; i < 120 * 400; i += 256) {
    int n = i / 400, k = i % 400;
    fw1q[n * 448 + k] = q8(fw1[i], fw1s);
  }
  for (int i = t; i < 10 * 120; i += 256) {
    int n = i / 120, k = i % 120;
    fw2q[n * 128 + k] = q8(fw2[i], fw2s);
  }
}

__global__ void k_quant_x(const float* __restrict__ x, int8_t* __restrict__ xq,
                          const float* __restrict__ scal, int n) {
  int i = blockIdx.x * blockDim.x + threadIdx.x;
  if (i >= n) return;
  float a0 = scal[IDX_A0];
  float q = fminf(fmaxf(rintf(x[i] / a0), -128.0f), 127.0f);
  xq[i] = (int8_t)(int)q;
}

__global__ void k_conv1(const int8_t* __restrict__ xq, const int8_t* __restrict__ w1q,
                        const float* __restrict__ b1, const float* __restrict__ scal,
                        float* __restrict__ pool1, float* __restrict__ maxout) {
  const int total = 8192 * 6 * 14 * 14;
  int idx = blockIdx.x * blockDim.x + threadIdx.x;
  float best = 0.0f;
  if (idx < total) {
    int px = idx % 14; int t = idx / 14;
    int py = t % 14; t /= 14;
    int oc = t % 6; int b = t / 6;
    const int8_t* xb = xq + (size_t)b * (3 * 32 * 32);
    const int8_t* wk = w1q + oc * 75;
    float s = scal[IDX_A0] * scal[IDX_W1S];
    float bq = bias_q(b1[oc], s);
    for (int dy = 0; dy < 2; ++dy)
      for (int dx = 0; dx < 2; ++dx) {
        int oy = 2 * py + dy, ox = 2 * px + dx;
        int acc = 0;
        for (int ic = 0; ic < 3; ++ic)
          for (int ky = 0; ky < 5; ++ky) {
            const int8_t* xr = xb + ic * 1024 + (oy + ky) * 32 + ox;
            const int8_t* wr = wk + ic * 25 + ky * 5;
            #pragma unroll
            for (int kx = 0; kx < 5; ++kx) acc += (int)xr[kx] * (int)wr[kx];
          }
        best = fmaxf(best, fmaxf(((float)acc + bq) * s, 0.0f));
      }
    pool1[idx] = best;
  }
  float w = wave_max(best);
  if ((threadIdx.x & 31) == 0) atomic_max_pos(maxout, w);
}

__global__ void k_conv2(const uint8_t* __restrict__ aq, const int8_t* __restrict__ w2q,
                        const float* __restrict__ b2, const float* __restrict__ scal,
                        float* __restrict__ pool2, float* __restrict__ maxout) {
  const int total = 8192 * 16 * 5 * 5;
  int idx = blockIdx.x * blockDim.x + threadIdx.x;
  float best = 0.0f;
  if (idx < total) {
    int px = idx % 5; int t = idx / 5;
    int py = t % 5; t /= 5;
    int oc = t % 16; int b = t / 16;
    const uint8_t* xb = aq + (size_t)b * (6 * 14 * 14);
    const int8_t* wk = w2q + oc * 150;
    float s1 = scal[IDX_MAX1] / 255.0f + 1e-12f;
    float s = s1 * scal[IDX_W2S];
    float bq = bias_q(b2[oc], s);
    for (int dy = 0; dy < 2; ++dy)
      for (int dx = 0; dx < 2; ++dx) {
        int oy = 2 * py + dy, ox = 2 * px + dx;
        int acc = 0;
        for (int ic = 0; ic < 6; ++ic)
          for (int ky = 0; ky < 5; ++ky) {
            const uint8_t* xr = xb + ic * 196 + (oy + ky) * 14 + ox;
            const int8_t* wr = wk + ic * 25 + ky * 5;
            #pragma unroll
            for (int kx = 0; kx < 5; ++kx) acc += (int)xr[kx] * (int)wr[kx];
          }
        best = fmaxf(best, fmaxf(((float)acc + bq) * s, 0.0f));
      }
    pool2[idx] = best;
  }
  float w = wave_max(best);
  if ((threadIdx.x & 31) == 0) atomic_max_pos(maxout, w);
}

__global__ void k_quant_u8(const float* __restrict__ in, uint8_t* __restrict__ out,
                           const float* __restrict__ maxp, int n) {
  int i = blockIdx.x * blockDim.x + threadIdx.x;
  if (i >= n) return;
  float s = maxp[0] / 255.0f + 1e-12f;
  float q = fminf(fmaxf(rintf(in[i] / s), 0.0f), 255.0f);
  out[i] = (uint8_t)(int)q;
}

__global__ void k_quant_act2(const float* __restrict__ in, uint8_t* __restrict__ out,
                             const float* __restrict__ scal) {
  int i = blockIdx.x * blockDim.x + threadIdx.x;
  if (i >= 8192 * 448) return;
  int n = i % 448; int m = i / 448;
  float s = scal[IDX_MAX2] / 255.0f + 1e-12f;
  float v = 0.0f;
  if (n < 400) {
    v = fminf(fmaxf(rintf(in[(size_t)m * 400 + n] / s), 0.0f), 255.0f);
  }
  out[i] = (uint8_t)(int)v;
}

__global__ void k_quant_fc1(const float* __restrict__ in, uint8_t* __restrict__ out,
                            const float* __restrict__ scal) {
  int i = blockIdx.x * blockDim.x + threadIdx.x;
  if (i >= 8192 * 128) return;
  int n = i & 127; int m = i >> 7;
  float s = scal[IDX_MAX3] / 255.0f + 1e-12f;
  float v = 0.0f;
  if (n < 120) {
    v = fminf(fmaxf(rintf(in[(size_t)m * 120 + n] / s), 0.0f), 255.0f);
  }
  out[i] = (uint8_t)(int)v;
}

static __device__ __forceinline__ v8i load_a_frag(const uint8_t* __restrict__ arow,
                                                  int k0, int kh) {
  v8i a;
  #pragma unroll
  for (int c = 0; c < 4; ++c) {
    int2 t = *reinterpret_cast<const int2*>(arow + k0 + (c >> 1) * 32 + kh * 16 + (c & 1) * 8);
    a[2 * c] = t.x; a[2 * c + 1] = t.y;
  }
  return a;
}
static __device__ __forceinline__ v8i load_b_frag(const int8_t* __restrict__ brow,
                                                  int k0, int kh) {
  v8i b;
  #pragma unroll
  for (int g = 0; g < 2; ++g) {
    int4 t = *reinterpret_cast<const int4*>(brow + k0 + g * 32 + kh * 16);
    b[4 * g] = t.x; b[4 * g + 1] = t.y; b[4 * g + 2] = t.z; b[4 * g + 3] = t.w;
  }
  return b;
}

__global__ __launch_bounds__(32)
void k_fc1_wmma(const uint8_t* __restrict__ A, const int8_t* __restrict__ B,
                const float* __restrict__ scal, const float* __restrict__ fb1,
                float* __restrict__ outf, float* __restrict__ maxout) {
  const int lane = threadIdx.x;
  const int m0 = blockIdx.x * 32;
  const int n0 = blockIdx.y * 16;
  const int ln = lane & 15;
  const int kh = lane >> 4;
  const uint8_t* arow0 = A + (size_t)(m0 + ln) * 448;
  const uint8_t* arow1 = A + (size_t)(m0 + 16 + ln) * 448;
  const int8_t*  brow  = B + (size_t)(n0 + ln) * 448;
  v8i acc0 = {}, acc1 = {};
  #pragma unroll
  for (int k0 = 0; k0 < 448; k0 += 64) {
    __builtin_prefetch(arow0 + k0 + 64, 0, 1);
    __builtin_prefetch(arow1 + k0 + 64, 0, 1);
    v8i a0 = load_a_frag(arow0, k0, kh);
    v8i a1 = load_a_frag(arow1, k0, kh);
    v8i bm = load_b_frag(brow, k0, kh);
    acc0 = __builtin_amdgcn_wmma_i32_16x16x64_iu8(false, a0, true, bm, acc0, false, false);
    acc1 = __builtin_amdgcn_wmma_i32_16x16x64_iu8(false, a1, true, bm, acc1, false, false);
  }
  const float s2 = scal[IDX_MAX2] / 255.0f + 1e-12f;
  const float s = s2 * scal[IDX_FW1S];
  float wmax = 0.0f;
  const int n = n0 + ln;
  if (n < 120) {
    float bq = bias_q(fb1[n], s);
    #pragma unroll
    for (int j = 0; j < 8; ++j) {
      int m = m0 + j + kh * 8;
      float v0 = fmaxf(((float)acc0[j] + bq) * s, 0.0f);
      float v1 = fmaxf(((float)acc1[j] + bq) * s, 0.0f);
      outf[(size_t)m * 120 + n] = v0;
      outf[(size_t)(m + 16) * 120 + n] = v1;
      wmax = fmaxf(wmax, fmaxf(v0, v1));
    }
  }
  wmax = wave_max(wmax);
  if (lane == 0) atomic_max_pos(maxout, wmax);
}

__global__ __launch_bounds__(32)
void k_fc2_wmma(const uint8_t* __restrict__ A, const int8_t* __restrict__ B,
                const float* __restrict__ scal, const float* __restrict__ fb2,
                float* __restrict__ out) {
  const int lane = threadIdx.x;
  const int m0 = blockIdx.x * 16;
  const int ln = lane & 15;
  const int kh = lane >> 4;
  const uint8_t* arow = A + (size_t)(m0 + ln) * 128;
  const int8_t*  brow = B + (size_t)ln * 128;
  v8i acc = {};
  #pragma unroll
  for (int k0 = 0; k0 < 128; k0 += 64) {
    v8i a  = load_a_frag(arow, k0, kh);
    v8i bm = load_b_frag(brow, k0, kh);
    acc = __builtin_amdgcn_wmma_i32_16x16x64_iu8(false, a, true, bm, acc, false, false);
  }
  const float s3 = scal[IDX_MAX3] / 255.0f + 1e-12f;
  const float s = s3 * scal[IDX_FW2S];
  const int n = ln;
  if (n < 10) {
    float bq = bias_q(fb2[n], s);
    #pragma unroll
    for (int j = 0; j < 8; ++j) {
      int m = m0 + j + kh * 8;
      out[(size_t)m * 10 + n] = ((float)acc[j] + bq) * s;
    }
  }
}

extern "C" void kernel_launch(void* const* d_in, const int* in_sizes, int n_in,
                              void* d_out, int out_size, void* d_ws, size_t ws_size,
                              hipStream_t stream) {
  (void)in_sizes; (void)n_in; (void)out_size; (void)ws_size;
  const float* x   = (const float*)d_in[0];
  const float* w1  = (const float*)d_in[1];
  const float* b1  = (const float*)d_in[2];
  const float* w2  = (const float*)d_in[3];
  const float* b2  = (const float*)d_in[4];
  const float* fw1 = (const float*)d_in[5];
  const float* fb1 = (const float*)d_in[6];
  const float* fw2 = (const float*)d_in[7];
  const float* fb2 = (const float*)d_in[8];
  float* out = (float*)d_out;

  char* ws = (char*)d_ws;
  float*   scal   = (float*)(ws + OFF_SCAL);
  int8_t*  w1q    = (int8_t*)(ws + OFF_W1Q);
  int8_t*  w2q    = (int8_t*)(ws + OFF_W2Q);
  int8_t*  fw1q   = (int8_t*)(ws + OFF_FW1Q);
  int8_t*  fw2q   = (int8_t*)(ws + OFF_FW2Q);
  int8_t*  xq     = (int8_t*)(ws + OFF_XQ);
  uint8_t* act1q  = (uint8_t*)(ws + OFF_ACT1Q);
  uint8_t* act2q  = (uint8_t*)(ws + OFF_ACT2Q);
  uint8_t* fc1q   = (uint8_t*)(ws + OFF_FC1Q);
  float*   pool1f = (float*)(ws + OFF_POOL1F);
  float*   pool2f = (float*)(ws + OFF_POOL2F);
  float*   fc1f   = (float*)(ws + OFF_FC1F);

  const int NX  = 8192 * 3 * 32 * 32;
  const int NP1 = 8192 * 6 * 14 * 14;
  const int NP2 = 8192 * 16 * 5 * 5;

  k_zero<<<(ZERO_BYTES / 4 + 255) / 256, 256, 0, stream>>>((float*)ws, ZERO_BYTES / 4);
  k_absmax<<<2048, 256, 0, stream>>>(x, NX, &scal[IDX_ABSMAX_X]);
  k_quant_weights<<<1, 256, 0, stream>>>(w1, w2, fw1, fw2, scal, w1q, w2q, fw1q, fw2q);
  k_quant_x<<<(NX + 255) / 256, 256, 0, stream>>>(x, xq, scal, NX);
  k_conv1<<<(NP1 + 255) / 256, 256, 0, stream>>>(xq, w1q, b1, scal, pool1f, &scal[IDX_MAX1]);
  k_quant_u8<<<(NP1 + 255) / 256, 256, 0, stream>>>(pool1f, act1q, &scal[IDX_MAX1], NP1);
  k_conv2<<<(NP2 + 255) / 256, 256, 0, stream>>>(act1q, w2q, b2, scal, pool2f, &scal[IDX_MAX2]);
  k_quant_act2<<<(8192 * 448 + 255) / 256, 256, 0, stream>>>(pool2f, act2q, scal);
  k_fc1_wmma<<<dim3(256, 8), 32, 0, stream>>>(act2q, fw1q, scal, fb1, fc1f, &scal[IDX_MAX3]);
  k_quant_fc1<<<(8192 * 128 + 255) / 256, 256, 0, stream>>>(fc1f, fc1q, scal);
  k_fc2_wmma<<<dim3(512, 1), 32, 0, stream>>>(fc1q, fw2q, scal, fb2, out);
}
